// Attention_70248485093458
// MI455X (gfx1250) — compile-verified, matrix operand order checked by reading, not run on hardware
//
#include <hip/hip_runtime.h>
#include <math.h>
#include <stdint.h>

typedef __attribute__((ext_vector_type(16))) _Float16 v16h;
typedef __attribute__((ext_vector_type(8)))  _Float16 v8h;
typedef __attribute__((ext_vector_type(8)))  float    v8f;

#define T_SEQ   2048
#define D_MODEL 256
#define N_QH    16
#define D_HEAD  256
#define WIN     512
#define EPSF    1.1920929e-07f
#define QDIM    4096   /* N_QH * D_HEAD */

// --------------------------------------------------------------------------
// Tensor Data Mover support (gfx1250). This toolchain exposes the 6-arg
// builtin (clang-23 / therock-10.0 headers):
//   (uint32x4 g0, int32x8 g1, int32x4 g2, int32x4 g3, int32x8 gx, i32 cpol)
// --------------------------------------------------------------------------
#ifndef USE_TDM
#if defined(__has_builtin)
#if __has_builtin(__builtin_amdgcn_tensor_load_to_lds) && \
    __has_builtin(__builtin_amdgcn_s_wait_tensorcnt)
#define USE_TDM 1
#endif
#endif
#endif
#ifndef USE_TDM
#define USE_TDM 0
#endif

#if USE_TDM
typedef __attribute__((ext_vector_type(4))) unsigned int u32x4;
typedef __attribute__((ext_vector_type(8))) int          i32x8;
typedef __attribute__((ext_vector_type(4))) int          i32x4;

// Issue a 2D f16 tile DMA: global (row-major, row_stride elements) -> LDS
// (contiguous tile_w per row). D# layout per CDNA5 ISA
// §8.3/8.4: group0 = {count/type/addr}, group1 = {sizes/strides}; groups 2/3
// zero (2D tensor). tensor_dim0/1 set to 1<<20 so no OOB clipping occurs.
__device__ __forceinline__ void tdm_load_2d_f16(const _Float16* gsrc,
                                                unsigned lds_byte_off,
                                                int tile_w, int tile_h,
                                                int row_stride_elems) {
  unsigned long long ga = (unsigned long long)(uintptr_t)gsrc;
  u32x4 g0 = { 1u,                                          // count=1 (valid user D#)
               lds_byte_off,                                // lds_addr
               (unsigned)ga,                                // global_addr[31:0]
               (unsigned)((ga >> 32) & 0x01FFFFFFu)         // global_addr[56:32]
                 | (2u << 30) };                            // type=2 ("image")
  i32x8 g1 = { (int)(1u << 16),                             // data_size=1 (2 bytes)
               0,                                           // atomic_barrier off
               0x10,                                        // tensor_dim0 = 1<<20 (hi bits)
               (int)(0x10u | ((unsigned)tile_w << 16)),     // tensor_dim1 hi | tile_dim0
               tile_h,                                      // tile_dim1 (tile_dim2 = 0)
               row_stride_elems,                            // tensor_dim0_stride[31:0]
               0, 0 };                                      // stride hi / dim1_stride
  i32x4 gz4 = { 0, 0, 0, 0 };
  i32x8 gz8 = { 0, 0, 0, 0, 0, 0, 0, 0 };
  __builtin_amdgcn_tensor_load_to_lds(g0, g1, gz4, gz4, gz8, 0);
}
#endif

// ---------------------------------------------------------------------------
// Fragment loader: 16x32 f16 tile, row-major source with leading dim `ld`.
// Per CDNA5 ISA (CDNA5 ISA, 16-bit A-matrix 16x32): lane L holds row (L&15);
// lanes 0-15 carry K=0..7 and 16..23, lanes 16-31 carry K=8..15 and 24..31.
// Both halves are 8 contiguous f16 => two 16-byte loads. B fragments (K x 16,
// mirrored layout with N in place of M) load identically from a [N,K] matrix.
// ---------------------------------------------------------------------------
__device__ __forceinline__ v16h load_frag(const _Float16* __restrict__ base, int ld) {
  const int lane = threadIdx.x & 31;
  const int r    = lane & 15;
  const int kh   = (lane >> 4) * 8;
  const _Float16* p = base + (size_t)r * ld + kh;
  v8h lo = *(const v8h*)(p);
  v8h hi = *(const v8h*)(p + 16);
  return __builtin_shufflevector(lo, hi, 0,1,2,3,4,5,6,7,8,9,10,11,12,13,14,15);
}

// ---------------------------------------------------------------------------
// Elementwise f32 -> f16
// ---------------------------------------------------------------------------
__global__ void f32_to_f16_kernel(const float* __restrict__ in,
                                  _Float16* __restrict__ out, int n) {
  int i = blockIdx.x * blockDim.x + threadIdx.x;
  int stride = gridDim.x * blockDim.x;
  for (; i < n; i += stride) out[i] = (_Float16)in[i];
}

// ---------------------------------------------------------------------------
// WMMA GEMM: C[M,N] (f32) = A[M,K] (f16, row-major) * B[N,K]^T (f16, row-major)
// Block = 256 threads = 8 waves; each wave owns a 32x32 tile; block = 128x64.
// M % 128 == 0, N % 64 == 0, K % 32 == 0 for all call sites here.
// (Direct global loads: all operands are L2-resident at this problem size;
//  round-1 codegen shows clause'd b128 loads pipelined against v_wmma.)
// ---------------------------------------------------------------------------
__global__ __launch_bounds__(256) void gemm_f16f32_kernel(
    const _Float16* __restrict__ A, const _Float16* __restrict__ B,
    float* __restrict__ C, int M, int N, int K) {
  const int wave = threadIdx.x >> 5;
  const int lane = threadIdx.x & 31;
  const int m0 = blockIdx.x * 128 + (wave >> 1) * 32;
  const int n0 = blockIdx.y * 64  + (wave & 1) * 32;

  v8f acc00 = {}, acc01 = {}, acc10 = {}, acc11 = {};
  for (int k0 = 0; k0 < K; k0 += 32) {
    v16h a0 = load_frag(A + (size_t)m0        * K + k0, K);
    v16h a1 = load_frag(A + (size_t)(m0 + 16) * K + k0, K);
    v16h b0 = load_frag(B + (size_t)n0        * K + k0, K);
    v16h b1 = load_frag(B + (size_t)(n0 + 16) * K + k0, K);
    acc00 = __builtin_amdgcn_wmma_f32_16x16x32_f16(false, a0, false, b0, (short)0, acc00, false, false);
    acc01 = __builtin_amdgcn_wmma_f32_16x16x32_f16(false, a0, false, b1, (short)0, acc01, false, false);
    acc10 = __builtin_amdgcn_wmma_f32_16x16x32_f16(false, a1, false, b0, (short)0, acc10, false, false);
    acc11 = __builtin_amdgcn_wmma_f32_16x16x32_f16(false, a1, false, b1, (short)0, acc11, false, false);
  }
  const int hi = lane >> 4, cn = lane & 15;
#pragma unroll
  for (int j = 0; j < 8; ++j) {
    int row = hi * 8 + j;
    C[(size_t)(m0 + row)      * N + n0      + cn] = acc00[j];
    C[(size_t)(m0 + row)      * N + n0 + 16 + cn] = acc01[j];
    C[(size_t)(m0 + 16 + row) * N + n0      + cn] = acc10[j];
    C[(size_t)(m0 + 16 + row) * N + n0 + 16 + cn] = acc11[j];
  }
}

// ---------------------------------------------------------------------------
// RoPE + RMSNorm. One block per timestep t, 288 threads = 18 groups of 16:
//   groups 0..15: the 16 q heads  -> qf[h][t][256] (f16)
//   group  16   : the single k head -> kf[t][256]  (f16)
//   group  17   : v pass-through, stored TRANSPOSED -> vt[d][t] (f16)
// ---------------------------------------------------------------------------
__global__ __launch_bounds__(288) void rope_norm_kernel(
    const float* __restrict__ qraw,   // [T, 4096]
    const float* __restrict__ kvraw,  // [T, 512]
    const float* __restrict__ qw, const float* __restrict__ kw,
    _Float16* __restrict__ qf,        // [16][T][256]
    _Float16* __restrict__ kf,        // [T][256]
    _Float16* __restrict__ vt)        // [256][T]
{
  const int t   = blockIdx.x;
  const int h   = threadIdx.x >> 4;
  const int l16 = threadIdx.x & 15;

  if (h < 17) {
    const float* src = (h < 16) ? (qraw + (size_t)t * QDIM + h * D_HEAD)
                                : (kvraw + (size_t)t * 512);
    const float* w   = (h < 16) ? qw : kw;
    _Float16* dst    = (h < 16) ? (qf + ((size_t)h * T_SEQ + t) * D_HEAD)
                                : (kf + (size_t)t * D_HEAD);
    float r1a[8], r2a[8];
    float ss = 0.f;
#pragma unroll
    for (int p = 0; p < 8; ++p) {
      int i = l16 + p * 16;                         // pair index 0..127
      float x1 = src[2 * i], x2 = src[2 * i + 1];
      // freqs = base^(-2*i/256) = exp(-ln(10000) * i / 128)
      float freq = __expf(-9.210340371976184f * (float)i * (1.0f / 128.0f));
      float ang  = (float)t * freq;
      float c = __cosf(ang), s = __sinf(ang);
      float r1 = x1 * c - x2 * s;
      float r2 = x1 * s + x2 * c;
      r1a[p] = r1; r2a[p] = r2;
      ss += r1 * r1 + r2 * r2;
    }
#pragma unroll
    for (int off = 1; off < 16; off <<= 1) ss += __shfl_xor(ss, off, 32);
    float inv = rsqrtf(ss * (1.0f / 256.0f) + EPSF);
#pragma unroll
    for (int p = 0; p < 8; ++p) {
      int i = l16 + p * 16;
      dst[2 * i]     = (_Float16)(r1a[p] * inv * w[2 * i]);
      dst[2 * i + 1] = (_Float16)(r2a[p] * inv * w[2 * i + 1]);
    }
  } else {
#pragma unroll
    for (int p = 0; p < 16; ++p) {
      int d = l16 + p * 16;
      vt[(size_t)d * T_SEQ + t] = (_Float16)kvraw[(size_t)t * 512 + 256 + d];
    }
  }
}

// ---------------------------------------------------------------------------
// Flash attention (sliding-window causal, window=512), one block per
// (head, 16-row q tile). Block = 128 threads = 4 waves; wave w owns the
// 64-wide d-slice [64w, 64w+64). Per 32-key chunk:
//   0) wave 0 issues a TDM 2D tile DMA: V slab (256 d-rows x 32 keys, strided
//      in [256][2048]) -> LDS; DMA overlaps the QK^T WMMAs and softmax
//   1) each wave: partial S (16x32) over its d-slice via 4 WMMAs
//   2) cross-wave sum of S through LDS (full 256-dim dot product)
//   3) online softmax (row max/sum via xor shuffles in 16-lane halves)
//   4) P (f16) staged through per-wave LDS to convert C-layout -> A-layout
//   5) s_wait_tensorcnt + barrier, then O += P @ V from the LDS V slab
// ---------------------------------------------------------------------------
__global__ __launch_bounds__(128) void attn_kernel(
    const _Float16* __restrict__ qf, const _Float16* __restrict__ kf,
    const _Float16* __restrict__ vt, _Float16* __restrict__ attn) {
  __shared__ float sred[4][2][256];
  __shared__ __align__(16) _Float16 pbuf[4][16][32];
  __shared__ __align__(16) _Float16 vbuf[256][32];   // V slab: [d][key]

  const int w    = threadIdx.x >> 5;
  const int lane = threadIdx.x & 31;
  const int hi   = lane >> 4;
  const int lo16 = lane & 15;
  const int q0   = blockIdx.x * 16;
  const int head = blockIdx.y;
  const int dw   = w * 64;

  const _Float16* qh = qf + (size_t)head * T_SEQ * D_HEAD;
  const v16h qA0 = load_frag(qh + (size_t)q0 * D_HEAD + dw,      D_HEAD);
  const v16h qA1 = load_frag(qh + (size_t)q0 * D_HEAD + dw + 32, D_HEAD);

  v8f O0 = {}, O1 = {}, O2 = {}, O3 = {};
  float m_run[8], l_run[8];
#pragma unroll
  for (int j = 0; j < 8; ++j) { m_run[j] = -1e30f; l_run[j] = 0.f; }

  int s_begin = q0 - WIN; if (s_begin < 0) s_begin = 0;
  s_begin &= ~31;

  for (int sb = s_begin; sb <= q0 + 15; sb += 32) {
    // --- 0) start the V slab DMA for this chunk (overlaps S + softmax) ---
#if USE_TDM
    if (threadIdx.x < 32) {
      // vt is [256][T_SEQ]; tile = 32 contiguous keys x 256 strided rows
      tdm_load_2d_f16(vt + sb, (unsigned)(uintptr_t)&vbuf[0][0],
                      /*tile_w=*/32, /*tile_h=*/256, /*stride=*/T_SEQ);
    }
#else
    // fallback: cooperative global->LDS copy, 16B per transfer
    {
      int tid = threadIdx.x;                 // 128 threads, 1024 v8h total
#pragma unroll
      for (int it = 0; it < 8; ++it) {
        int idx = tid * 8 + it;              // v8h index
        int row = idx >> 2, col = (idx & 3) * 8;
        *(v8h*)&vbuf[row][col] = *(const v8h*)(vt + (size_t)row * T_SEQ + sb + col);
      }
    }
#endif

    // --- 1) partial S over this wave's 64-dim slice, two 16-key sub-tiles ---
    v8f c0 = {}, c1 = {};
    {
      v16h kB;
      kB = load_frag(kf + (size_t)sb * D_HEAD + dw, D_HEAD);
      c0 = __builtin_amdgcn_wmma_f32_16x16x32_f16(false, qA0, false, kB, (short)0, c0, false, false);
      kB = load_frag(kf + (size_t)sb * D_HEAD + dw + 32, D_HEAD);
      c0 = __builtin_amdgcn_wmma_f32_16x16x32_f16(false, qA1, false, kB, (short)0, c0, false, false);
      kB = load_frag(kf + (size_t)(sb + 16) * D_HEAD + dw, D_HEAD);
      c1 = __builtin_amdgcn_wmma_f32_16x16x32_f16(false, qA0, false, kB, (short)0, c1, false, false);
      kB = load_frag(kf + (size_t)(sb + 16) * D_HEAD + dw + 32, D_HEAD);
      c1 = __builtin_amdgcn_wmma_f32_16x16x32_f16(false, qA1, false, kB, (short)0, c1, false, false);
    }
#pragma unroll
    for (int j = 0; j < 8; ++j) {
      sred[w][0][j * 32 + lane] = c0[j];
      sred[w][1][j * 32 + lane] = c1[j];
    }
    __syncthreads();

    // --- 2) full 256-dim S = sum of the 4 per-wave partials ---
    float sv0[8], sv1[8], a8[8];
    bool  ok0[8], ok1[8];
#pragma unroll
    for (int j = 0; j < 8; ++j) {
      float s0 = sred[0][0][j * 32 + lane] + sred[1][0][j * 32 + lane] +
                 sred[2][0][j * 32 + lane] + sred[3][0][j * 32 + lane];
      float s1 = sred[0][1][j * 32 + lane] + sred[1][1][j * 32 + lane] +
                 sred[2][1][j * 32 + lane] + sred[3][1][j * 32 + lane];
      int qrow = q0 + hi * 8 + j;
      int si0  = sb + lo16;
      int si1  = sb + 16 + lo16;
      ok0[j] = (si0 <= qrow) && (qrow - si0 <= WIN);
      ok1[j] = (si1 <= qrow) && (qrow - si1 <= WIN);
      sv0[j] = s0 * 0.0625f;   // 1/sqrt(256)
      sv1[j] = s1 * 0.0625f;
    }
    __syncthreads();   // sred reusable next chunk

    // --- 3) online softmax: row stats via shuffles within 16-lane halves ---
#pragma unroll
    for (int j = 0; j < 8; ++j) {
      float rm = fmaxf(ok0[j] ? sv0[j] : -1e30f, ok1[j] ? sv1[j] : -1e30f);
#pragma unroll
      for (int off = 1; off < 16; off <<= 1) rm = fmaxf(rm, __shfl_xor(rm, off, 32));
      float mn    = fmaxf(m_run[j], rm);
      float alpha = __expf(m_run[j] - mn);
      m_run[j] = mn;
      float p0 = ok0[j] ? __expf(sv0[j] - mn) : 0.f;   // explicit zero: avoids
      float p1 = ok1[j] ? __expf(sv1[j] - mn) : 0.f;   // exp(0)=1 on all-masked rows
      float rs = p0 + p1;
#pragma unroll
      for (int off = 1; off < 16; off <<= 1) rs += __shfl_xor(rs, off, 32);
      l_run[j] = l_run[j] * alpha + rs;
      a8[j] = alpha;
      int row = hi * 8 + j;
      pbuf[w][row][lo16]      = (_Float16)p0;
      pbuf[w][row][16 + lo16] = (_Float16)p1;
    }
#pragma unroll
    for (int j = 0; j < 8; ++j) {
      O0[j] *= a8[j]; O1[j] *= a8[j]; O2[j] *= a8[j]; O3[j] *= a8[j];
    }

    // --- 4) per-wave LDS staging is in-order within a wave; wait, reload P ---
    asm volatile("s_wait_dscnt 0" ::: "memory");
    v16h pA;
    {
      const _Float16* pp = &pbuf[w][lo16][hi * 8];
      v8h plo = *(const v8h*)(pp);
      v8h phi = *(const v8h*)(pp + 16);
      pA = __builtin_shufflevector(plo, phi, 0,1,2,3,4,5,6,7,8,9,10,11,12,13,14,15);
    }

    // --- 5) V slab ready: TENSORcnt drain on the issuing wave, then barrier ---
#if USE_TDM
    if (threadIdx.x < 32) __builtin_amdgcn_s_wait_tensorcnt(0);
#endif
    __syncthreads();

    // O += P @ V from LDS; B-fragments are contiguous 16B rows of vbuf
    {
      const _Float16* vp0 = &vbuf[dw +  0 + lo16][hi * 8];
      const _Float16* vp1 = &vbuf[dw + 16 + lo16][hi * 8];
      const _Float16* vp2 = &vbuf[dw + 32 + lo16][hi * 8];
      const _Float16* vp3 = &vbuf[dw + 48 + lo16][hi * 8];
      v16h vB0 = __builtin_shufflevector(*(const v8h*)vp0, *(const v8h*)(vp0 + 16),
                                         0,1,2,3,4,5,6,7,8,9,10,11,12,13,14,15);
      v16h vB1 = __builtin_shufflevector(*(const v8h*)vp1, *(const v8h*)(vp1 + 16),
                                         0,1,2,3,4,5,6,7,8,9,10,11,12,13,14,15);
      v16h vB2 = __builtin_shufflevector(*(const v8h*)vp2, *(const v8h*)(vp2 + 16),
                                         0,1,2,3,4,5,6,7,8,9,10,11,12,13,14,15);
      v16h vB3 = __builtin_shufflevector(*(const v8h*)vp3, *(const v8h*)(vp3 + 16),
                                         0,1,2,3,4,5,6,7,8,9,10,11,12,13,14,15);
      O0 = __builtin_amdgcn_wmma_f32_16x16x32_f16(false, pA, false, vB0, (short)0, O0, false, false);
      O1 = __builtin_amdgcn_wmma_f32_16x16x32_f16(false, pA, false, vB1, (short)0, O1, false, false);
      O2 = __builtin_amdgcn_wmma_f32_16x16x32_f16(false, pA, false, vB2, (short)0, O2, false, false);
      O3 = __builtin_amdgcn_wmma_f32_16x16x32_f16(false, pA, false, vB3, (short)0, O3, false, false);
    }
    __syncthreads();   // vbuf reads complete before next chunk's DMA
  }

  // --- finalize: O / l, write f16 to [T][4096] for the output projection ---
#pragma unroll
  for (int j = 0; j < 8; ++j) {
    float inv = 1.0f / l_run[j];
    int trow  = q0 + hi * 8 + j;
    _Float16* dst = attn + (size_t)trow * QDIM + head * D_HEAD + dw;
    dst[ 0 + lo16] = (_Float16)(O0[j] * inv);
    dst[16 + lo16] = (_Float16)(O1[j] * inv);
    dst[32 + lo16] = (_Float16)(O2[j] * inv);
    dst[48 + lo16] = (_Float16)(O3[j] * inv);
  }
}

// ---------------------------------------------------------------------------
extern "C" void kernel_launch(void* const* d_in, const int* in_sizes, int n_in,
                              void* d_out, int out_size, void* d_ws, size_t ws_size,
                              hipStream_t stream) {
  (void)in_sizes; (void)n_in; (void)out_size; (void)ws_size;
  const float* x   = (const float*)d_in[0];
  const float* wq  = (const float*)d_in[1];
  const float* wkv = (const float*)d_in[2];
  const float* wo  = (const float*)d_in[3];
  const float* qnw = (const float*)d_in[4];
  const float* knw = (const float*)d_in[5];
  float* out = (float*)d_out;

  char* ws = (char*)d_ws;
  size_t off = 0;
  auto carve = [&](size_t bytes) -> void* {
    void* p = ws + off;
    off += (bytes + 255) & ~(size_t)255;
    return p;
  };
  _Float16* x16    = (_Float16*)carve((size_t)T_SEQ * D_MODEL * 2);
  _Float16* wq16   = (_Float16*)carve((size_t)QDIM * D_MODEL * 2);
  _Float16* wkv16  = (_Float16*)carve((size_t)512 * D_MODEL * 2);
  _Float16* wo16   = (_Float16*)carve((size_t)D_MODEL * QDIM * 2);
  _Float16* qf16   = (_Float16*)carve((size_t)N_QH * T_SEQ * D_HEAD * 2);
  _Float16* kf16   = (_Float16*)carve((size_t)T_SEQ * D_HEAD * 2);
  _Float16* vt16   = (_Float16*)carve((size_t)D_HEAD * T_SEQ * 2);
  // qraw/kvraw are dead after rope_norm; alias attn16 over the same region.
  float*    qraw   = (float*)carve((size_t)T_SEQ * QDIM * 4);
  float*    kvraw  = (float*)carve((size_t)T_SEQ * 512 * 4);
  _Float16* attn16 = (_Float16*)qraw;

  // 1) convert inputs to f16
  f32_to_f16_kernel<<<dim3(512),  256, 0, stream>>>(x,   x16,   T_SEQ * D_MODEL);
  f32_to_f16_kernel<<<dim3(1024), 256, 0, stream>>>(wq,  wq16,  QDIM * D_MODEL);
  f32_to_f16_kernel<<<dim3(128),  256, 0, stream>>>(wkv, wkv16, 512 * D_MODEL);
  f32_to_f16_kernel<<<dim3(1024), 256, 0, stream>>>(wo,  wo16,  D_MODEL * QDIM);

  // 2) projections: q = x @ wq^T, kv = x @ wkv^T (WMMA)
  gemm_f16f32_kernel<<<dim3(16, 64), 256, 0, stream>>>(x16, wq16,  qraw,  T_SEQ, QDIM, D_MODEL);
  gemm_f16f32_kernel<<<dim3(16, 8),  256, 0, stream>>>(x16, wkv16, kvraw, T_SEQ, 512,  D_MODEL);

  // 3) RoPE + RMSNorm -> f16 q/k, transposed f16 v
  rope_norm_kernel<<<dim3(T_SEQ), 288, 0, stream>>>(qraw, kvraw, qnw, knw, qf16, kf16, vt16);

  // 4) sliding-window flash attention (WMMA QK^T and PV, TDM-staged V)
  attn_kernel<<<dim3(T_SEQ / 16, N_QH), 128, 0, stream>>>(qf16, kf16, vt16, attn16);

  // 5) output projection: out = attn @ wo^T (WMMA)
  gemm_f16f32_kernel<<<dim3(16, 4), 256, 0, stream>>>(attn16, wo16, out, T_SEQ, D_MODEL, QDIM);
}
